// GraphAutoEncoder_85899345978
// MI455X (gfx1250) — matrix operand order repaired and checked by reading, not compiled, not run on hardware
//
#include <hip/hip_runtime.h>
#include <hip/hip_bf16.h>
#include <stdint.h>

#define DEVI static __device__ __forceinline__

typedef __attribute__((ext_vector_type(16))) __bf16 v16bf;
typedef __attribute__((ext_vector_type(8)))  float  v8f;

#define NN 50000
#define NE 1600000

// ---------------- fragment helpers (CDNA5 WMMA 16x16x32 bf16) ----------------

union Frag { uint32_t u[8]; v16bf v; };

DEVI uint16_t bf16bits(float x) {
  union { __bf16 b; uint16_t s; } u;
  u.b = (__bf16)x;
  return u.s;
}

DEVI uint32_t pk2(float lo, float hi) {
  return (uint32_t)bf16bits(lo) | ((uint32_t)bf16bits(hi) << 16);
}

// A-matrix local K index for VGPR v, lane-half h (ISA 7.12.2, 16-bit A 16x32)
DEVI int kA(int v, int h) { return (v < 4) ? (2 * v + 8 * h) : (8 + 2 * v + 8 * h); }

DEVI v8f splat8(float x) {
  v8f r;
#pragma unroll
  for (int i = 0; i < 8; ++i) r[i] = x;
  return r;
}

DEVI v8f wmma(const Frag& a, const Frag& b, v8f c) {
  return __builtin_amdgcn_wmma_f32_16x16x32_bf16(false, a.v, false, b.v,
                                                 (short)0, c, false, false);
}

// B fragment from swizzled pool: [tile][v][lane] uint32, coalesced per v
DEVI Frag ldB(const uint32_t* __restrict__ pool, int tile, int lane) {
  Frag f;
  const uint32_t* p = pool + tile * 256 + lane;
#pragma unroll
  for (int v = 0; v < 8; ++v) f.u[v] = p[v * 32];
  return f;
}

// A fragment from row-major f32 matrix (row fixed per lane), zero-pad cols >= ncols
DEVI Frag ldA_f32(const float* m, int row, int ld, int cb, int h, int ncols) {
  Frag f;
#pragma unroll
  for (int v = 0; v < 8; ++v) {
    int k = kA(v, h);
    float lo = (k < ncols)     ? m[row * ld + cb + k]     : 0.f;
    float hi = (k + 1 < ncols) ? m[row * ld + cb + k + 1] : 0.f;
    f.u[v] = pk2(lo, hi);
  }
  return f;
}

// A fragment of (a + b), full 32 cols valid
DEVI Frag ldA_sum(const float* a, const float* b, int row, int ld, int cb, int h) {
  Frag f;
#pragma unroll
  for (int v = 0; v < 8; ++v) {
    int k = kA(v, h);
    int i0 = row * ld + cb + k;
    f.u[v] = pk2(a[i0] + b[i0], a[i0 + 1] + b[i0 + 1]);
  }
  return f;
}

// A fragment from LDS bf16 tile [16][Np] row-major
DEVI Frag ldA_lds(const uint16_t* s, int Np, int r, int cb, int h) {
  Frag f;
#pragma unroll
  for (int v = 0; v < 8; ++v) {
    int k = cb + kA(v, h);
    f.u[v] = *(const uint32_t*)(s + r * Np + k);   // k even, 4B aligned
  }
  return f;
}

// Store D tile (optionally relu'd) as bf16 into LDS tile [16][Np]
DEVI void stD_lds(uint16_t* s, int Np, int r, int h, int col0, v8f d, bool relu) {
#pragma unroll
  for (int v = 0; v < 8; ++v) {
    float x = d[v];
    if (relu) x = fmaxf(x, 0.f);
    s[(v + 8 * h) * Np + col0 + r] = bf16bits(x);
  }
}

// wave-local LDS ordering (no workgroup barrier needed; per-wave staging buffers)
DEVI void lds_fence() {
  __builtin_amdgcn_wave_barrier();
  asm volatile("s_wait_dscnt 0" ::: "memory");
  __builtin_amdgcn_wave_barrier();
}

// ---------------- weight-fragment pool layout (tiles of 256 uint32) ----------------

enum { T_NODE = 0, T_EDGE = 4, T_L1_0 = 8, T_L2_0 = 16, T_L1_1 = 24, T_L2_1 = 32,
       T_ENC = 40, T_DEC1 = 44, T_DEC2 = 52, T_MLP1 = 54, T_MLP2 = 66, T_MLP3 = 70,
       T_TOTAL = 71 };
enum { O_NODE = 0, O_EDGE = 64, O_L1_0 = 128, O_L2_0 = 192, O_L1_1 = 256, O_L2_1 = 320,
       O_ENC = 384, O_DEC1 = 416, O_DEC2 = 480, O_MLP1 = 496, O_MLP2 = 560, O_MLP3 = 592,
       O_TOTAL = 608 };

struct WInfo { const float* w; const float* b; int K, N, KS, NT, foff, boff; };
struct PrepArgs { WInfo wi[12]; };

// Swizzle f32 weights -> bf16 B-fragments + zero-padded f32 biases.
__global__ void k_prep(PrepArgs pa, uint32_t* __restrict__ pool, float* __restrict__ bias) {
  int tid = blockIdx.x * blockDim.x + threadIdx.x;
  int stride = gridDim.x * blockDim.x;
  for (int i = 0; i < 12; ++i) {
    WInfo wi = pa.wi[i];
    int total = wi.KS * wi.NT * 256;
    for (int e = tid; e < total; e += stride) {
      int tile = e >> 8, v = (e >> 5) & 7, lane = e & 31;
      int nt = tile / wi.KS, ks = tile % wi.KS;      // tile = nt*KS + ks
      int n = nt * 16 + (lane & 15), h = lane >> 4;
      int k0 = ks * 32 + kA(v, h);                   // B layout: same k order as A (kA)
      float lo = (k0 < wi.K && n < wi.N)     ? wi.w[k0 * wi.N + n]       : 0.f;
      float hi = (k0 + 1 < wi.K && n < wi.N) ? wi.w[(k0 + 1) * wi.N + n] : 0.f;
      pool[wi.foff * 256 + e] = pk2(lo, hi);
    }
    int npad = wi.NT * 16;
    for (int j = tid; j < npad; j += stride)
      bias[wi.boff + j] = (j < wi.N) ? wi.b[j] : 0.f;
  }
}

// ---------------- h = x @ Wnode + bnode  (K=32, N=64) ----------------
__global__ void k_node(const float* __restrict__ x, const uint32_t* __restrict__ pool,
                       const float* __restrict__ bias, float* __restrict__ h) {
  int lane = threadIdx.x & 31, wave = threadIdx.x >> 5;
  int tile = blockIdx.x * 8 + wave;
  if (tile >= NN / 16) return;                       // wave-uniform
  int r = lane & 15, hh = lane >> 4;
  int base = tile * 16;
  Frag a = ldA_f32(x, base + r, 32, 0, hh, 32);
#pragma unroll
  for (int nt = 0; nt < 4; ++nt) {
    Frag b = ldB(pool, T_NODE + nt, lane);
    v8f acc = splat8(bias[O_NODE + nt * 16 + r]);
    acc = wmma(a, b, acc);
#pragma unroll
    for (int v = 0; v < 8; ++v)
      h[(base + v + 8 * hh) * 64 + nt * 16 + r] = acc[v];
  }
}

// ---------------- scatter: e = edge_emb(attr); msg = relu(h[src]+e); agg[dst] += msg ----------------
__global__ void k_scatter(const int* __restrict__ ei, const float* __restrict__ ea,
                          const uint32_t* __restrict__ pool, const float* __restrict__ bias,
                          const float* __restrict__ h, float* __restrict__ agg) {
  int lane = threadIdx.x & 31, wave = threadIdx.x >> 5;
  int tile = blockIdx.x * 8 + wave;
  if (tile >= NE / 16) return;
  int r = lane & 15, hh = lane >> 4;
  int ebase = tile * 16;
  const int* src = ei;
  const int* dst = ei + NE;
  Frag a = ldA_f32(ea, ebase + r, 16, 0, hh, 16);    // K=16 zero-padded to 32
#pragma unroll
  for (int nt = 0; nt < 4; ++nt) {
    Frag b = ldB(pool, T_EDGE + nt, lane);
    v8f acc = splat8(bias[O_EDGE + nt * 16 + r]);
    acc = wmma(a, b, acc);
#pragma unroll
    for (int v = 0; v < 8; ++v) {
      int row = ebase + v + 8 * hh;
      int s = src[row], d = dst[row];
      float val = acc[v] + h[s * 64 + nt * 16 + r];
      val = fmaxf(val, 0.f);
      unsafeAtomicAdd(&agg[d * 64 + nt * 16 + r], val);
    }
  }
}

// ---------------- GINE mlp: g2 = lin2(relu(lin1(agg+h))); accumulate BN sums ----------------
__global__ void k_gine(const uint32_t* __restrict__ pool, const float* __restrict__ bias,
                       const float* __restrict__ h, const float* __restrict__ agg,
                       float* __restrict__ g2, float* __restrict__ bnsum,
                       int f1, int b1, int f2, int b2) {
  __shared__ uint16_t smem[8][16 * 64];
  int lane = threadIdx.x & 31, wave = threadIdx.x >> 5;
  int tile = blockIdx.x * 8 + wave;
  if (tile >= NN / 16) return;
  int r = lane & 15, hh = lane >> 4;
  int base = tile * 16;
  uint16_t* s = smem[wave];

  Frag a0 = ldA_sum(agg, h, base + r, 64, 0, hh);
  Frag a1 = ldA_sum(agg, h, base + r, 64, 32, hh);
#pragma unroll
  for (int nt = 0; nt < 4; ++nt) {                   // lin1 + relu -> LDS
    v8f acc = splat8(bias[b1 + nt * 16 + r]);
    acc = wmma(a0, ldB(pool, f1 + nt * 2 + 0, lane), acc);
    acc = wmma(a1, ldB(pool, f1 + nt * 2 + 1, lane), acc);
    stD_lds(s, 64, r, hh, nt * 16, acc, true);
  }
  lds_fence();
  Frag c0 = ldA_lds(s, 64, r, 0, hh);
  Frag c1 = ldA_lds(s, 64, r, 32, hh);
#pragma unroll
  for (int nt = 0; nt < 4; ++nt) {                   // lin2 -> g2, BN sums
    v8f acc = splat8(bias[b2 + nt * 16 + r]);
    acc = wmma(c0, ldB(pool, f2 + nt * 2 + 0, lane), acc);
    acc = wmma(c1, ldB(pool, f2 + nt * 2 + 1, lane), acc);
    float s1 = 0.f, s2 = 0.f;
#pragma unroll
    for (int v = 0; v < 8; ++v) {
      float xv = acc[v];
      g2[(base + v + 8 * hh) * 64 + nt * 16 + r] = xv;
      s1 += xv;
      s2 += xv * xv;
    }
    unsafeAtomicAdd(&bnsum[nt * 16 + r], s1);
    unsafeAtomicAdd(&bnsum[64 + nt * 16 + r], s2);
  }
}

// ---------------- BN apply + residual: h = 0.5*(h + relu(bn(g2))) ----------------
__global__ void k_bn(const float* __restrict__ g2, const float* __restrict__ bnsum,
                     const float* __restrict__ gamma, const float* __restrict__ beta,
                     float* __restrict__ h) {
  int i = blockIdx.x * blockDim.x + threadIdx.x;
  if (i >= NN * 64) return;
  int n = i & 63;
  float mean = bnsum[n] * (1.f / NN);
  float var = bnsum[64 + n] * (1.f / NN) - mean * mean;
  float g = (g2[i] - mean) * rsqrtf(var + 1e-5f) * gamma[n] + beta[n];
  h[i] = 0.5f * (h[i] + fmaxf(g, 0.f));
}

// ---------------- z = h @ Wenc + benc (K=64, N=32) ----------------
__global__ void k_enc(const uint32_t* __restrict__ pool, const float* __restrict__ bias,
                      const float* __restrict__ h, float* __restrict__ z) {
  int lane = threadIdx.x & 31, wave = threadIdx.x >> 5;
  int tile = blockIdx.x * 8 + wave;
  if (tile >= NN / 16) return;
  int r = lane & 15, hh = lane >> 4;
  int base = tile * 16;
  Frag a0 = ldA_f32(h, base + r, 64, 0, hh, 32);
  Frag a1 = ldA_f32(h, base + r, 64, 32, hh, 32);
#pragma unroll
  for (int nt = 0; nt < 2; ++nt) {
    v8f acc = splat8(bias[O_ENC + nt * 16 + r]);
    acc = wmma(a0, ldB(pool, T_ENC + nt * 2 + 0, lane), acc);
    acc = wmma(a1, ldB(pool, T_ENC + nt * 2 + 1, lane), acc);
#pragma unroll
    for (int v = 0; v < 8; ++v)
      z[(base + v + 8 * hh) * 32 + nt * 16 + r] = acc[v];
  }
}

// ---------------- fused edge head: dec1/dec2/recon-err/mlp1/mlp2/mlp3 ----------------
__global__ void k_edge(const int* __restrict__ ei, const float* __restrict__ ea,
                       const float* __restrict__ z, const uint32_t* __restrict__ pool,
                       const float* __restrict__ bias, float* __restrict__ out) {
  __shared__ uint16_t smem[8][16 * 64];
  int lane = threadIdx.x & 31, wave = threadIdx.x >> 5;
  int tile = blockIdx.x * 8 + wave;
  if (tile >= NE / 16) return;
  int r = lane & 15, hh = lane >> 4;
  int ebase = tile * 16;
  const int* src = ei;
  const int* dst = ei + NE;
  uint16_t* s = smem[wave];

  int sA = src[ebase + r], dA = dst[ebase + r];
  Frag e0 = ldA_f32(z, sA, 32, 0, hh, 32);           // edge_input cols 0..31 = z[src]
  Frag e1 = ldA_f32(z, dA, 32, 0, hh, 32);           // cols 32..63 = z[dst]

  // dec1 + relu -> LDS
#pragma unroll
  for (int nt = 0; nt < 4; ++nt) {
    v8f acc = splat8(bias[O_DEC1 + nt * 16 + r]);
    acc = wmma(e0, ldB(pool, T_DEC1 + nt * 2 + 0, lane), acc);
    acc = wmma(e1, ldB(pool, T_DEC1 + nt * 2 + 1, lane), acc);
    stD_lds(s, 64, r, hh, nt * 16, acc, true);
  }
  lds_fence();
  Frag u0 = ldA_lds(s, 64, r, 0, hh);
  Frag u1 = ldA_lds(s, 64, r, 32, hh);
  lds_fence();

  // dec2 -> rec (16x16 D tile)
  v8f rec = splat8(bias[O_DEC2 + r]);
  rec = wmma(u0, ldB(pool, T_DEC2 + 0, lane), rec);
  rec = wmma(u1, ldB(pool, T_DEC2 + 1, lane), rec);

  // recon error: mean over 16 dims of (rec - attr)^2, via per-wave LDS row reduce
  float* sf = (float*)s;                             // reuse staging (1KB of 2KB)
#pragma unroll
  for (int v = 0; v < 8; ++v) {
    float av = ea[(ebase + v + 8 * hh) * 16 + r];
    float df = rec[v] - av;
    sf[(v + 8 * hh) * 16 + r] = df * df;
  }
  lds_fence();
  float err = 0.f;
  if (lane < 16) {
    for (int j = 0; j < 16; ++j) err += sf[lane * 16 + j];
    err *= (1.f / 16.f);
  }
  // feat ks2 fragment: local cols 0..15 = attr, col16 = err (feat dim 80), rest 0
  Frag f2;
#pragma unroll
  for (int v = 0; v < 4; ++v) {
    int k = kA(v, hh);                               // 0..15
    f2.u[v] = pk2(ea[(ebase + r) * 16 + k], ea[(ebase + r) * 16 + k + 1]);
  }
  f2.u[4] = (hh == 0) ? pk2(err, 0.f) : 0u;
  f2.u[5] = 0u; f2.u[6] = 0u; f2.u[7] = 0u;
  lds_fence();                                       // err reads done before reuse

  // mlp1 (K=96 padded, N=64 padded) + relu -> LDS
#pragma unroll
  for (int nt = 0; nt < 4; ++nt) {
    v8f acc = splat8(bias[O_MLP1 + nt * 16 + r]);
    acc = wmma(e0, ldB(pool, T_MLP1 + nt * 3 + 0, lane), acc);
    acc = wmma(e1, ldB(pool, T_MLP1 + nt * 3 + 1, lane), acc);
    acc = wmma(f2, ldB(pool, T_MLP1 + nt * 3 + 2, lane), acc);
    stD_lds(s, 64, r, hh, nt * 16, acc, true);
  }
  lds_fence();
  Frag m0 = ldA_lds(s, 64, r, 0, hh);
  Frag m1 = ldA_lds(s, 64, r, 32, hh);
  lds_fence();

  // mlp2 (K=64 padded, N=32 padded) + relu -> LDS
#pragma unroll
  for (int nt = 0; nt < 2; ++nt) {
    v8f acc = splat8(bias[O_MLP2 + nt * 16 + r]);
    acc = wmma(m0, ldB(pool, T_MLP2 + nt * 2 + 0, lane), acc);
    acc = wmma(m1, ldB(pool, T_MLP2 + nt * 2 + 1, lane), acc);
    stD_lds(s, 32, r, hh, nt * 16, acc, true);
  }
  lds_fence();
  Frag p0 = ldA_lds(s, 32, r, 0, hh);

  // mlp3 (K=32 padded, N=16 padded; real N=2)
  v8f o = splat8(bias[O_MLP3 + r]);
  o = wmma(p0, ldB(pool, T_MLP3, lane), o);
  if (r < 2) {
#pragma unroll
    for (int v = 0; v < 8; ++v)
      out[(ebase + v + 8 * hh) * 2 + r] = o[v];
  }
}

// ---------------- host launcher ----------------

extern "C" void kernel_launch(void* const* d_in, const int* in_sizes, int n_in,
                              void* d_out, int out_size, void* d_ws, size_t ws_size,
                              hipStream_t stream) {
  (void)in_sizes; (void)n_in; (void)out_size; (void)ws_size;
  const float* x  = (const float*)d_in[0];
  const int*   ei = (const int*)d_in[1];
  const float* ea = (const float*)d_in[2];
  // params flattened in JAX pytree (sorted-dict-key) leaf order:
  const float* bn_beta[2]  = { (const float*)d_in[3], (const float*)d_in[5] };
  const float* bn_gamma[2] = { (const float*)d_in[4], (const float*)d_in[6] };
  const float* l1b[2] = { (const float*)d_in[7],  (const float*)d_in[11] };
  const float* l1w[2] = { (const float*)d_in[8],  (const float*)d_in[12] };
  const float* l2b[2] = { (const float*)d_in[9],  (const float*)d_in[13] };
  const float* l2w[2] = { (const float*)d_in[10], (const float*)d_in[14] };
  const float* dec1b = (const float*)d_in[15]; const float* dec1w = (const float*)d_in[16];
  const float* dec2b = (const float*)d_in[17]; const float* dec2w = (const float*)d_in[18];
  const float* edgb  = (const float*)d_in[19]; const float* edgw  = (const float*)d_in[20];
  const float* encb  = (const float*)d_in[21]; const float* encw  = (const float*)d_in[22];
  const float* m1b   = (const float*)d_in[23]; const float* m1w   = (const float*)d_in[24];
  const float* m2b   = (const float*)d_in[25]; const float* m2w   = (const float*)d_in[26];
  const float* m3b   = (const float*)d_in[27]; const float* m3w   = (const float*)d_in[28];
  const float* nodb  = (const float*)d_in[29]; const float* nodw  = (const float*)d_in[30];

  char* w = (char*)d_ws;
  uint32_t* pool = (uint32_t*)w;  w += (size_t)T_TOTAL * 256 * 4;
  float* bias    = (float*)w;     w += 640 * 4;
  float* hbuf    = (float*)w;     w += (size_t)NN * 64 * 4;
  float* g2      = (float*)w;     w += (size_t)NN * 64 * 4;
  float* agg     = (float*)w;     w += (size_t)NN * 64 * 4;
  float* zbuf    = (float*)w;     w += (size_t)NN * 32 * 4;
  float* bnsum   = (float*)w;     w += 128 * 4;

  PrepArgs pa = { {
    { nodw,  nodb,  32, 64, 1, 4, T_NODE, O_NODE },
    { edgw,  edgb,  16, 64, 1, 4, T_EDGE, O_EDGE },
    { l1w[0], l1b[0], 64, 64, 2, 4, T_L1_0, O_L1_0 },
    { l2w[0], l2b[0], 64, 64, 2, 4, T_L2_0, O_L2_0 },
    { l1w[1], l1b[1], 64, 64, 2, 4, T_L1_1, O_L1_1 },
    { l2w[1], l2b[1], 64, 64, 2, 4, T_L2_1, O_L2_1 },
    { encw,  encb,  64, 32, 2, 2, T_ENC,  O_ENC },
    { dec1w, dec1b, 64, 64, 2, 4, T_DEC1, O_DEC1 },
    { dec2w, dec2b, 64, 16, 2, 1, T_DEC2, O_DEC2 },
    { m1w,   m1b,   81, 50, 3, 4, T_MLP1, O_MLP1 },
    { m2w,   m2b,   50, 25, 2, 2, T_MLP2, O_MLP2 },
    { m3w,   m3b,   25,  2, 1, 1, T_MLP3, O_MLP3 },
  } };

  const int NODE_BLOCKS = (NN / 16 + 7) / 8;   // 391
  const int EDGE_BLOCKS = (NE / 16) / 8;       // 12500

  k_prep<<<64, 256, 0, stream>>>(pa, pool, bias);
  k_node<<<NODE_BLOCKS, 256, 0, stream>>>(x, pool, bias, hbuf);

  for (int L = 0; L < 2; ++L) {
    hipMemsetAsync(agg, 0, (size_t)NN * 64 * 4, stream);
    hipMemsetAsync(bnsum, 0, 128 * 4, stream);
    k_scatter<<<EDGE_BLOCKS, 256, 0, stream>>>(ei, ea, pool, bias, hbuf, agg);
    int f1 = L ? T_L1_1 : T_L1_0, b1 = L ? O_L1_1 : O_L1_0;
    int f2 = L ? T_L2_1 : T_L2_0, b2 = L ? O_L2_1 : O_L2_0;
    k_gine<<<NODE_BLOCKS, 256, 0, stream>>>(pool, bias, hbuf, agg, g2, bnsum, f1, b1, f2, b2);
    k_bn<<<(NN * 64 + 255) / 256, 256, 0, stream>>>(g2, bnsum, bn_gamma[L], bn_beta[L], hbuf);
  }

  k_enc<<<NODE_BLOCKS, 256, 0, stream>>>(pool, bias, hbuf, zbuf);
  k_edge<<<EDGE_BLOCKS, 256, 0, stream>>>(ei, ea, zbuf, pool, bias, (float*)d_out);
}
